// Hier_Matcher_9388798509739
// MI455X (gfx1250) — matrix operand order repaired and checked by reading, not compiled, not run on hardware
//
#include <hip/hip_runtime.h>
#include <hip/hip_bf16.h>

typedef __attribute__((ext_vector_type(16))) __bf16 v16bf;
typedef __attribute__((ext_vector_type(8)))  __bf16 v8bf;
typedef __attribute__((ext_vector_type(8)))  float  v8f;

#define DD 256
#define KMAXC 4
#define KVSPLIT 32

__device__ __forceinline__ int padMT(int m) { return (m + 31) & ~31; }

// ---------------- elementwise / setup ----------------

__global__ void init_full(const float* __restrict__ d2, const float* __restrict__ d3,
                          int n2elems, int total, float* __restrict__ full,
                          float* __restrict__ dout) {
  int i = blockIdx.x * 256 + threadIdx.x;
  if (i >= total) return;
  float v = (i < n2elems) ? d2[i] : d3[i - n2elems];
  full[i] = v;
  dout[i] = v;
}

// f32 (rows x cols, row-major) -> bf16 transposed (cols x rows). Tiled via LDS.
__global__ void f2bf_tr(const float* __restrict__ s, __bf16* __restrict__ d,
                        int rows, int cols) {
  __shared__ __bf16 t[32][33];
  int c0 = blockIdx.x * 32, r0 = blockIdx.y * 32;
  for (int i = threadIdx.y; i < 32; i += 8) {
    int r = r0 + i, c = c0 + threadIdx.x;
    t[i][threadIdx.x] = (r < rows && c < cols) ? (__bf16)s[(size_t)r * cols + c]
                                               : (__bf16)0.f;
  }
  __syncthreads();
  for (int i = threadIdx.y; i < 32; i += 8) {
    int c = c0 + i, r = r0 + threadIdx.x;
    if (c < cols && r < rows) d[(size_t)c * rows + r] = t[threadIdx.x][i];
  }
}

// elu(x)+1 fused with transpose: x (Mp x 256 f32) -> y (256 x Mp bf16)
__global__ void elu_tr(const float* __restrict__ x, __bf16* __restrict__ y,
                       const int* __restrict__ Mdev) {
  int Mp = padMT(*Mdev);
  int r0 = blockIdx.y * 32;
  if (r0 >= Mp) return;
  int c0 = blockIdx.x * 32;
  __shared__ __bf16 t[32][33];
  for (int i = threadIdx.y; i < 32; i += 8) {
    int r = r0 + i, c = c0 + threadIdx.x;
    float v = (r < Mp) ? x[(size_t)r * DD + c] : 0.f;
    float e = (v > 0.f) ? v + 1.f : __expf(v);
    t[i][threadIdx.x] = (__bf16)e;
  }
  __syncthreads();
  for (int i = threadIdx.y; i < 32; i += 8) {
    int c = c0 + i, r = r0 + threadIdx.x;
    if (r < Mp) y[(size_t)c * Mp + r] = t[threadIdx.x][i];
  }
}

// ---------------- kmeans ----------------

__global__ void kinit(const float* __restrict__ X, float* __restrict__ C,
                      float* __restrict__ cnorm) {
  int k = blockIdx.x, d = threadIdx.x;
  float v = X[(size_t)k * DD + d];
  C[k * DD + d] = v;
  __shared__ float sh[DD];
  sh[d] = v * v;
  __syncthreads();
  for (int s = 128; s > 0; s >>= 1) { if (d < s) sh[d] += sh[d + s]; __syncthreads(); }
  if (d == 0) cnorm[k] = sh[0];
}

__global__ void kassign(const float* __restrict__ X, const float* __restrict__ C,
                        const float* __restrict__ cnorm, int K, int Nconst,
                        const int* __restrict__ Ndev, int* __restrict__ labels,
                        float* __restrict__ psum, float* __restrict__ pcnt, int accum) {
  int tid = threadIdx.x;
  int N = Ndev ? *Ndev : Nconst;
  __shared__ float cL[KMAXC * DD];
  __shared__ int clsL[256];
  for (int i = tid; i < K * DD; i += 256) cL[i] = C[i];
  __syncthreads();
  int n = blockIdx.x * 256 + tid;
  int cl = 0;
  if (n < N) {
    float dot0 = 0.f, dot1 = 0.f, dot2 = 0.f, dot3 = 0.f;
    const float* xr = X + (size_t)n * DD;
    for (int d = 0; d < DD; ++d) {
      float xv = xr[d];
      dot0 += xv * cL[d];
      if (K > 1) dot1 += xv * cL[DD + d];
      if (K > 2) dot2 += xv * cL[2 * DD + d];
      if (K > 3) dot3 += xv * cL[3 * DD + d];
    }
    float best = cnorm[0] - 2.f * dot0; cl = 0;
    if (K > 1) { float s = cnorm[1] - 2.f * dot1; if (s < best) { best = s; cl = 1; } }
    if (K > 2) { float s = cnorm[2] - 2.f * dot2; if (s < best) { best = s; cl = 2; } }
    if (K > 3) { float s = cnorm[3] - 2.f * dot3; if (s < best) { best = s; cl = 3; } }
    if (labels) labels[n] = cl;
  }
  if (!accum) return;  // uniform across block
  clsL[tid] = (n < N) ? cl : -1;
  __syncthreads();
  // deterministic per-block column accumulation (thread = dim)
  int d = tid;
  float a0 = 0.f, a1 = 0.f, a2 = 0.f, a3 = 0.f;
  int base = blockIdx.x * 256;
  for (int j = 0; j < 256; ++j) {
    int c = clsL[j];
    if (c < 0) continue;
    float xv = X[(size_t)(base + j) * DD + d];
    if (c == 0) a0 += xv; else if (c == 1) a1 += xv;
    else if (c == 2) a2 += xv; else a3 += xv;
  }
  size_t pb = (size_t)blockIdx.x * (size_t)(K * DD);
  psum[pb + d] = a0;
  if (K > 1) psum[pb + DD + d] = a1;
  if (K > 2) psum[pb + 2 * DD + d] = a2;
  if (K > 3) psum[pb + 3 * DD + d] = a3;
  if (tid < K) {
    int c = 0;
    for (int j = 0; j < 256; ++j) if (clsL[j] == tid) ++c;
    pcnt[(size_t)blockIdx.x * K + tid] = (float)c;
  }
}

__global__ void kreduce(const float* __restrict__ psum, const float* __restrict__ pcnt,
                        int K, int nblocks, float* __restrict__ sums,
                        float* __restrict__ cnts) {
  int k = blockIdx.x, d = threadIdx.x;
  float s = 0.f;
  for (int b = 0; b < nblocks; ++b) s += psum[(size_t)b * (K * DD) + k * DD + d];
  sums[k * DD + d] = s;
  if (d == 0) {
    float c = 0.f;
    for (int b = 0; b < nblocks; ++b) c += pcnt[(size_t)b * K + k];
    cnts[k] = c;
  }
}

__global__ void kupdate(float* __restrict__ C, float* __restrict__ cnorm,
                        const float* __restrict__ sums, const float* __restrict__ cnts) {
  int k = blockIdx.x, d = threadIdx.x;
  float cnt = cnts[k];
  float v = (cnt > 0.f) ? sums[k * DD + d] / fmaxf(cnt, 1.f) : C[k * DD + d];
  C[k * DD + d] = v;
  __shared__ float sh[DD];
  sh[d] = v * v;
  __syncthreads();
  for (int s = 128; s > 0; s >>= 1) { if (d < s) sh[d] += sh[d + s]; __syncthreads(); }
  if (d == 0) cnorm[k] = sh[0];
}

// ---------------- ordered compaction ----------------

__global__ void ccount(const int* __restrict__ lab, int target, int Nconst,
                       const int* __restrict__ Ndev, int* __restrict__ bcnt) {
  int N = Ndev ? *Ndev : Nconst;
  int n = blockIdx.x * 256 + threadIdx.x;
  int p = (n < N && lab[n] == target) ? 1 : 0;
  __shared__ int sh[256];
  sh[threadIdx.x] = p;
  __syncthreads();
  for (int s = 128; s > 0; s >>= 1) {
    if (threadIdx.x < s) sh[threadIdx.x] += sh[threadIdx.x + s];
    __syncthreads();
  }
  if (threadIdx.x == 0) bcnt[blockIdx.x] = sh[0];
}

__global__ void cscan(const int* __restrict__ bcnt, int nb, int* __restrict__ boff,
                      int* __restrict__ Mout) {
  __shared__ int sh[256];
  int t = threadIdx.x;
  int v = (t < nb) ? bcnt[t] : 0;
  sh[t] = v;
  __syncthreads();
  for (int off = 1; off < 256; off <<= 1) {
    int add = (t >= off) ? sh[t - off] : 0;
    __syncthreads();
    sh[t] += add;
    __syncthreads();
  }
  if (t < nb) boff[t] = sh[t] - v;
  if (t == 255) *Mout = sh[255];
}

__global__ void cwrite(const int* __restrict__ lab, int target, int Nconst,
                       const int* __restrict__ Ndev, const int* __restrict__ boff,
                       int* __restrict__ outIdx) {
  int N = Ndev ? *Ndev : Nconst;
  int t = threadIdx.x;
  int n = blockIdx.x * 256 + t;
  int p = (n < N && lab[n] == target) ? 1 : 0;
  __shared__ int sh[256];
  sh[t] = p;
  __syncthreads();
  for (int off = 1; off < 256; off <<= 1) {
    int add = (t >= off) ? sh[t - off] : 0;
    __syncthreads();
    sh[t] += add;
    __syncthreads();
  }
  if (p) outIdx[boff[blockIdx.x] + sh[t] - 1] = n;
}

__global__ void gather_rows(const float* __restrict__ src, const int* __restrict__ idx,
                            const int* __restrict__ Mdev, float* __restrict__ xs,
                            __bf16* __restrict__ cat) {
  int m = blockIdx.x;
  int M = *Mdev, Mp = padMT(M);
  if (m >= Mp) return;
  int d = threadIdx.x;
  float v = 0.f;
  if (m < M) v = src[(size_t)idx[m] * DD + d];
  xs[(size_t)m * DD + d] = v;
  cat[(size_t)m * 512 + d] = (__bf16)v;
}

// ---------------- WMMA GEMMs ----------------
// C(M x N) = A(M x K, bf16 row-major, lda) * B(K x N) with B stored TRANSPOSED
// (Bt is N x K row-major -> all fragment loads contiguous b128 pairs).
// Each wave computes an MT x NT grid of 16x16 tiles (register-level reuse of
// both A and B fragments). Rows padded to 32 so MT=2 never straddles the edge.
template <int MT, int NT>
__global__ void __launch_bounds__(128)
gemm_bf16(const __bf16* __restrict__ A, int lda, long long az,
          const __bf16* __restrict__ Bt, long long bz,
          float* __restrict__ Cf, __bf16* __restrict__ Cb, int ldc, long long cz,
          int trCb, const int* __restrict__ Mdev, int K, int relu) {
  int M = *Mdev, Mp = padMT(M);
  int row0 = blockIdx.y * (16 * MT);
  if (row0 >= Mp) return;
  int wave = threadIdx.x >> 5, lane = threadIdx.x & 31;
  int col0 = (blockIdx.x * 4 + wave) * (16 * NT);
  A += (size_t)blockIdx.z * az;
  Bt += (size_t)blockIdx.z * bz;
  v8f acc[MT][NT];
#pragma unroll
  for (int mt = 0; mt < MT; ++mt)
#pragma unroll
    for (int t = 0; t < NT; ++t) acc[mt][t] = (v8f){};
  int mrow = lane & 15;
  int kh = (lane >> 4) * 16;  // A: K-half per lane group
  int bk = (lane >> 4) * 16;  // B: K-half per lane group
  const __bf16* ap0 = A + (size_t)(row0 + mrow) * lda + kh;
  const __bf16* bp0 = Bt + (size_t)(col0 + (lane & 15)) * K + bk;
  for (int kk = 0; kk < K; kk += 32) {
    v16bf a[MT], b[NT];
#pragma unroll
    for (int mt = 0; mt < MT; ++mt) {
      const __bf16* ap = ap0 + (size_t)mt * 16 * lda + kk;
      v8bf alo = *(const v8bf*)(ap);
      v8bf ahi = *(const v8bf*)(ap + 8);
      a[mt] = __builtin_shufflevector(alo, ahi, 0, 1, 2, 3, 4, 5, 6, 7, 8, 9, 10, 11,
                                      12, 13, 14, 15);
    }
#pragma unroll
    for (int t = 0; t < NT; ++t)
      b[t] = *(const v16bf*)(bp0 + (size_t)t * 16 * K + kk);
#pragma unroll
    for (int mt = 0; mt < MT; ++mt)
#pragma unroll
      for (int t = 0; t < NT; ++t)
        acc[mt][t] = __builtin_amdgcn_wmma_f32_16x16x32_bf16(false, a[mt], false, b[t],
                                                             (short)0, acc[mt][t],
                                                             false, false);
  }
#pragma unroll
  for (int mt = 0; mt < MT; ++mt) {
    int r0 = row0 + mt * 16 + ((lane >> 4) ? 8 : 0);
#pragma unroll
    for (int t = 0; t < NT; ++t) {
      int cn = col0 + t * 16 + (lane & 15);
      if (trCb) {  // bf16 output stored transposed (N x Mp): contiguous v8bf store
        v8bf st;
#pragma unroll
        for (int v = 0; v < 8; ++v) {
          float val = acc[mt][t][v];
          if (relu) val = fmaxf(val, 0.f);
          st[v] = (__bf16)val;
        }
        *(v8bf*)(Cb + (size_t)cn * Mp + r0) = st;
      } else {
#pragma unroll
        for (int v = 0; v < 8; ++v) {
          float val = acc[mt][t][v];
          if (relu) val = fmaxf(val, 0.f);
          size_t o = (size_t)blockIdx.z * cz + (size_t)(r0 + v) * ldc + cn;
          if (Cf) Cf[o] = val;
          if (Cb) Cb[o] = (__bf16)val;
        }
      }
    }
  }
}

// Split-K partial: Cpart[split][h][r,c] = sum_{n in chunk} At[(h*64+r)*Mp+n]*Bt[(h*64+c)*Mp+n]
// With At=Vt, Bt=Kt this yields KV^T = V^T K per head (KV = K^T V).
// grid (KVSPLIT, 4, H), 4 waves = 4 col tiles. Deterministic (fixed chunking).
__global__ void __launch_bounds__(128)
gemm_kv(const __bf16* __restrict__ At, const __bf16* __restrict__ Bt,
        float* __restrict__ Cpart, const int* __restrict__ Mdev) {
  int Mp = padMT(*Mdev);
  int h = blockIdx.z;
  int split = blockIdx.x;
  int iters = Mp >> 5;
  int per = (iters + KVSPLIT - 1) / KVSPLIT;
  int k0 = split * per * 32;
  int k1 = k0 + per * 32;
  if (k1 > Mp) k1 = Mp;
  int wave = threadIdx.x >> 5, lane = threadIdx.x & 31;
  int r0t = blockIdx.y * 16;
  int c0t = wave * 16;
  v8f acc = {};
  int rr = r0t + (lane & 15);
  int kh = (lane >> 4) * 16;
  int bk = (lane >> 4) * 16;
  int cc = c0t + (lane & 15);
  const __bf16* ap0 = At + (size_t)(h * 64 + rr) * Mp + kh;
  const __bf16* bp0 = Bt + (size_t)(h * 64 + cc) * Mp + bk;
  for (int kk = k0; kk < k1; kk += 32) {
    v8bf alo = *(const v8bf*)(ap0 + kk);
    v8bf ahi = *(const v8bf*)(ap0 + kk + 8);
    v16bf a = __builtin_shufflevector(alo, ahi, 0, 1, 2, 3, 4, 5, 6, 7, 8, 9, 10, 11,
                                      12, 13, 14, 15);
    v16bf b = *(const v16bf*)(bp0 + kk);
    acc = __builtin_amdgcn_wmma_f32_16x16x32_bf16(false, a, false, b, (short)0, acc,
                                                  false, false);
  }
  int cn = c0t + (lane & 15);
  int r0 = r0t + ((lane >> 4) ? 8 : 0);
#pragma unroll
  for (int v = 0; v < 8; ++v)
    Cpart[(size_t)split * 16384 + (size_t)h * 4096 + (size_t)(r0 + v) * 64 + cn] =
        acc[v];
}

// Fixed-order reduction of split-K partials -> bf16 KV^T (Bt layout for Q@KV GEMM)
__global__ void kvreduce(const float* __restrict__ Cpart, __bf16* __restrict__ KVbt,
                         int total) {
  int i = blockIdx.x * 256 + threadIdx.x;
  if (i >= total) return;
  float s = 0.f;
  for (int p = 0; p < KVSPLIT; ++p) s += Cpart[(size_t)p * 16384 + i];
  KVbt[i] = (__bf16)s;
}

// ---------------- layer epilogues ----------------

__global__ void elu1(const float* __restrict__ x, __bf16* __restrict__ y,
                     const int* __restrict__ Mdev) {
  int i = blockIdx.x * 256 + threadIdx.x;
  int Mp = padMT(*Mdev);
  if (i >= Mp * DD) return;
  float v = x[i];
  float r = (v > 0.f) ? (v + 1.f) : __expf(v);
  y[i] = (__bf16)r;
}

// Ksum[d] = sum_{n<M} Kt[d*Mp + n]  (Kt transposed -> coalesced)
__global__ void ksum_red(const __bf16* __restrict__ Kt, const int* __restrict__ Mdev,
                         float* __restrict__ Ks) {
  int d = blockIdx.x, t = threadIdx.x;
  int M = *Mdev, Mp = padMT(M);
  float s = 0.f;
  for (int n = t; n < M; n += 256) s += (float)Kt[(size_t)d * Mp + n];
  __shared__ float sh[256];
  sh[t] = s;
  __syncthreads();
  for (int o = 128; o > 0; o >>= 1) { if (t < o) sh[t] += sh[t + o]; __syncthreads(); }
  if (t == 0) Ks[d] = sh[0];
}

__global__ void zscale(const __bf16* __restrict__ Qb, const float* __restrict__ Ks,
                       const float* __restrict__ P, __bf16* __restrict__ msg,
                       const int* __restrict__ Mdev) {
  int m = blockIdx.x;
  int M = *Mdev, Mp = padMT(M);
  if (m >= Mp) return;
  int d = threadIdx.x;
  __shared__ float sh[256];
  float q = (float)Qb[(size_t)m * DD + d];
  sh[d] = q * Ks[d];
  __syncthreads();
  for (int o = 32; o > 0; o >>= 1) {
    if ((d & 63) < o) sh[d] += sh[d + o];
    __syncthreads();
  }
  float z = 1.f / (sh[(d >> 6) << 6] + 1e-6f);
  float out = (m < M) ? P[(size_t)m * DD + d] * z : 0.f;
  msg[(size_t)m * DD + d] = (__bf16)out;
}

__global__ void layernorm_k(const float* __restrict__ in, const float* __restrict__ g,
                            const float* __restrict__ b, const float* __restrict__ res,
                            float* __restrict__ outf, __bf16* __restrict__ outb,
                            int ldbo, const int* __restrict__ Mdev) {
  int m = blockIdx.x;
  int M = *Mdev, Mp = padMT(M);
  if (m >= Mp) return;
  int d = threadIdx.x;
  __shared__ float sh[256];
  float v = (m < M) ? in[(size_t)m * DD + d] : 0.f;
  sh[d] = v;
  __syncthreads();
  for (int o = 128; o > 0; o >>= 1) { if (d < o) sh[d] += sh[d + o]; __syncthreads(); }
  float mu = sh[0] * (1.f / 256.f);
  __syncthreads();
  float c = v - mu;
  sh[d] = c * c;
  __syncthreads();
  for (int o = 128; o > 0; o >>= 1) { if (d < o) sh[d] += sh[d + o]; __syncthreads(); }
  float var = sh[0] * (1.f / 256.f);
  float y = c * rsqrtf(var + 1e-5f) * g[d] + b[d];
  if (res) y += res[(size_t)m * DD + d];
  if (m >= M) y = 0.f;
  if (outf) outf[(size_t)m * DD + d] = y;
  if (outb) outb[(size_t)m * ldbo + d] = (__bf16)y;
}

__global__ void scatter_out(const float* __restrict__ y2, const int* __restrict__ idx1,
                            const int* __restrict__ idx2, const int* __restrict__ M2dev,
                            const int* __restrict__ M1dev, float* __restrict__ dout) {
  int m = blockIdx.x;
  if (m >= *M2dev) return;
  if (*M1dev <= 20) return;  // cnt1 > 20 gate from reference
  int gidx = idx1[idx2[m]];
  dout[(size_t)gidx * DD + threadIdx.x] = y2[(size_t)m * DD + threadIdx.x];
}

// ---------------- host ----------------

extern "C" void kernel_launch(void* const* d_in, const int* in_sizes, int n_in,
                              void* d_out, int out_size, void* d_ws, size_t ws_size,
                              hipStream_t stream) {
  (void)n_in; (void)out_size; (void)ws_size;
  const float* desc2d = (const float*)d_in[0];
  const float* desc3d = (const float*)d_in[1];
  const float* l1w[10];
  const float* l2w[10];
  for (int i = 0; i < 10; ++i) { l1w[i] = (const float*)d_in[2 + i]; l2w[i] = (const float*)d_in[12 + i]; }
  const int n2 = in_sizes[0] / DD;
  const int n3 = in_sizes[1] / DD;
  const int N = n2 + n3;
  const int nblk = (N + 255) / 256;  // <= 256 required by cscan
  const int gy2 = (N + 31) / 32;
  float* dout = (float*)d_out;

  char* base = (char*)d_ws;
  size_t off = 0;
  auto alloc = [&](size_t bytes) -> void* {
    void* p = base + off;
    off = (off + bytes + 255) & ~(size_t)255;
    return p;
  };
  size_t fb = (size_t)N * DD * sizeof(float);
  float*  full = (float*)alloc(fb);
  float*  xs   = (float*)alloc(fb);
  float*  qf   = (float*)alloc(fb);   // reused as P (Q@KV) after elu
  float*  kf   = (float*)alloc(fb);   // reused as Wm output
  float*  w2o  = (float*)alloc(fb);
  float*  y1s  = (float*)alloc(fb);
  float*  y2s  = (float*)alloc(fb);
  __bf16* catb = (__bf16*)alloc((size_t)N * 512 * 2);
  __bf16* h1b  = (__bf16*)alloc((size_t)N * 512 * 2);
  __bf16* Qb   = (__bf16*)alloc((size_t)N * DD * 2);
  __bf16* Kbt  = (__bf16*)alloc((size_t)N * DD * 2);  // transposed (256 x Mp)
  __bf16* Vbt  = (__bf16*)alloc((size_t)N * DD * 2);  // transposed (256 x Mp)
  __bf16* msgb = (__bf16*)alloc((size_t)N * DD * 2);
  float*  KVpart = (float*)alloc((size_t)KVSPLIT * 16384 * sizeof(float));
  __bf16* KVbt = (__bf16*)alloc(4 * 64 * 64 * 2);
  float*  Ksum = (float*)alloc(DD * sizeof(float));
  float*  Cc   = (float*)alloc(KMAXC * DD * sizeof(float));
  float*  cnorm= (float*)alloc(KMAXC * sizeof(float));
  float*  sums = (float*)alloc(KMAXC * DD * sizeof(float));
  float*  cnts = (float*)alloc(KMAXC * sizeof(float));
  float*  psum = (float*)alloc((size_t)nblk * KMAXC * DD * sizeof(float));
  float*  pcnt = (float*)alloc((size_t)nblk * KMAXC * sizeof(float));
  int*    cls1 = (int*)alloc((size_t)N * sizeof(int));
  int*    cls2 = (int*)alloc((size_t)N * sizeof(int));
  int*    idx1 = (int*)alloc((size_t)N * sizeof(int));
  int*    idx2 = (int*)alloc((size_t)N * sizeof(int));
  int*    bcnt = (int*)alloc(256 * sizeof(int));
  int*    boff = (int*)alloc(256 * sizeof(int));
  int*    Mv   = (int*)alloc(2 * sizeof(int));
  int* M1dev = Mv;
  int* M2dev = Mv + 1;

  // bf16 TRANSPOSED weight copies: Wq,Wk,Wv,Wm,W1,W2 (stored N x K)
  const int wrows[6] = {DD, DD, DD, DD, 512, 512};
  const int wcols[6] = {DD, DD, DD, DD, 512, DD};
  __bf16* Wt[2][6];
  for (int L = 0; L < 2; ++L)
    for (int wi = 0; wi < 6; ++wi) {
      Wt[L][wi] = (__bf16*)alloc((size_t)wrows[wi] * wcols[wi] * 2);
      const float* src = L ? l2w[wi] : l1w[wi];
      f2bf_tr<<<dim3(wcols[wi] / 32, wrows[wi] / 32), dim3(32, 8), 0, stream>>>(
          src, Wt[L][wi], wrows[wi], wcols[wi]);
    }

  int totElem = N * DD;
  init_full<<<(totElem + 255) / 256, 256, 0, stream>>>(desc2d, desc3d, n2 * DD, totElem,
                                                       full, dout);

  // LoFTR linear-attention layer on compact rows [0, M): cat[:, :256] holds bf16 x.
  auto run_layer = [&](int L, const int* Mdev, float* yout) {
    const float *g1 = L ? l2w[6] : l1w[6], *b1 = L ? l2w[7] : l1w[7];
    const float *g2 = L ? l2w[8] : l1w[8], *b2 = L ? l2w[9] : l1w[9];
    // q,k,v projections (A = cat with lda=512 -> first 256 cols = x)
    gemm_bf16<2, 2><<<dim3(2, gy2, 1), 128, 0, stream>>>(catb, 512, 0, Wt[L][0], 0,
                                                         qf, nullptr, DD, 0, 0, Mdev, DD, 0);
    gemm_bf16<2, 2><<<dim3(2, gy2, 1), 128, 0, stream>>>(catb, 512, 0, Wt[L][1], 0,
                                                         kf, nullptr, DD, 0, 0, Mdev, DD, 0);
    gemm_bf16<2, 2><<<dim3(2, gy2, 1), 128, 0, stream>>>(catb, 512, 0, Wt[L][2], 0,
                                                         nullptr, Vbt, 0, 0, 1, Mdev, DD, 0);
    int eblk = (N * DD + 255) / 256;
    elu1<<<eblk, 256, 0, stream>>>(qf, Qb, Mdev);
    elu_tr<<<dim3(8, (N + 31) / 32), dim3(32, 8), 0, stream>>>(kf, Kbt, Mdev);
    ksum_red<<<DD, 256, 0, stream>>>(Kbt, Mdev, Ksum);
    gemm_kv<<<dim3(KVSPLIT, 4, 4), 128, 0, stream>>>(Vbt, Kbt, KVpart, Mdev);  // = KV^T
    kvreduce<<<64, 256, 0, stream>>>(KVpart, KVbt, 16384);
    // P_h = Q_h @ KV_h, batched over heads (z); B supplied as KV^T (Bt convention)
    gemm_bf16<2, 1><<<dim3(1, gy2, 4), 128, 0, stream>>>(Qb, DD, 64, KVbt, 4096,
                                                         qf, nullptr, DD, 64, 0, Mdev, 64, 0);
    zscale<<<N, 256, 0, stream>>>(Qb, Ksum, qf, msgb, Mdev);
    gemm_bf16<2, 2><<<dim3(2, gy2, 1), 128, 0, stream>>>(msgb, DD, 0, Wt[L][3], 0,
                                                         kf, nullptr, DD, 0, 0, Mdev, DD, 0);
    layernorm_k<<<N, 256, 0, stream>>>(kf, g1, b1, nullptr, nullptr, catb + 256, 512,
                                       Mdev);
    gemm_bf16<2, 2><<<dim3(4, gy2, 1), 128, 0, stream>>>(catb, 512, 0, Wt[L][4], 0,
                                                         nullptr, h1b, 512, 0, 0, Mdev, 512, 1);
    gemm_bf16<2, 2><<<dim3(2, gy2, 1), 128, 0, stream>>>(h1b, 512, 0, Wt[L][5], 0,
                                                         w2o, nullptr, DD, 0, 0, Mdev, 512, 0);
    layernorm_k<<<N, 256, 0, stream>>>(w2o, g2, b2, xs, yout, nullptr, 0, Mdev);
  };

  // ---- level-1 kmeans (K=4, all points, init = first 4 rows) ----
  kinit<<<4, 256, 0, stream>>>(full, Cc, cnorm);
  for (int it = 0; it < 10; ++it) {
    kassign<<<nblk, 256, 0, stream>>>(full, Cc, cnorm, 4, N, nullptr, cls1, psum, pcnt, 1);
    kreduce<<<4, 256, 0, stream>>>(psum, pcnt, 4, nblk, sums, cnts);
    kupdate<<<4, 256, 0, stream>>>(Cc, cnorm, sums, cnts);
  }
  kassign<<<nblk, 256, 0, stream>>>(full, Cc, cnorm, 4, N, nullptr, cls1, nullptr,
                                    nullptr, 0);

  for (int ci = 0; ci < 4; ++ci) {
    ccount<<<nblk, 256, 0, stream>>>(cls1, ci, N, nullptr, bcnt);
    cscan<<<1, 256, 0, stream>>>(bcnt, nblk, boff, M1dev);
    cwrite<<<nblk, 256, 0, stream>>>(cls1, ci, N, nullptr, boff, idx1);
    gather_rows<<<N, 256, 0, stream>>>(full, idx1, M1dev, xs, catb);
    run_layer(0, M1dev, y1s);

    // level-2 kmeans (K=2) on compact y1 rows; init = first 2 masked points
    kinit<<<2, 256, 0, stream>>>(y1s, Cc, cnorm);
    for (int it = 0; it < 10; ++it) {
      kassign<<<nblk, 256, 0, stream>>>(y1s, Cc, cnorm, 2, N, M1dev, cls2, psum, pcnt, 1);
      kreduce<<<2, 256, 0, stream>>>(psum, pcnt, 2, nblk, sums, cnts);
      kupdate<<<2, 256, 0, stream>>>(Cc, cnorm, sums, cnts);
    }
    kassign<<<nblk, 256, 0, stream>>>(y1s, Cc, cnorm, 2, N, M1dev, cls2, nullptr,
                                      nullptr, 0);

    for (int cj = 0; cj < 2; ++cj) {
      ccount<<<nblk, 256, 0, stream>>>(cls2, cj, N, M1dev, bcnt);
      cscan<<<1, 256, 0, stream>>>(bcnt, nblk, boff, M2dev);
      cwrite<<<nblk, 256, 0, stream>>>(cls2, cj, N, M1dev, boff, idx2);
      gather_rows<<<N, 256, 0, stream>>>(y1s, idx2, M2dev, xs, catb);
      run_layer(1, M2dev, y2s);
      scatter_out<<<N, 256, 0, stream>>>(y2s, idx1, idx2, M2dev, M1dev, dout);
    }
  }
}
